// VoxelGNNDiscriminator_76570676953621
// MI455X (gfx1250) — matrix operand order repaired and checked by reading, not compiled, not run on hardware
//
#include <hip/hip_runtime.h>
#include <hip/hip_bf16.h>
#include <math.h>

typedef __bf16 bf16;
typedef __attribute__((ext_vector_type(16))) __bf16 v16bf;
typedef __attribute__((ext_vector_type(8)))  __bf16 v8bf;
typedef __attribute__((ext_vector_type(8)))  float  v8f;

#define NN      200000
#define NE      1600000
#define LDIM    128
#define VDIM    64
#define NCLS    8
#define INDIM   200
#define INPAD   224
#define HID     128
#define EPS_LN  1e-5f

// ---------------------------------------------------------------------------
// WMMA GEMM: Out[M, NT*16] = act(A[M,K] @ Wt^T + bias)
//   A  : bf16 row-major, leading dim lda (K padded to 32)
//   Wt : bf16 row-major [Nout][K]  (i.e. W transposed), leading dim ldw = K
//   MODE 0: f32 dense store, no bias, no activation (pre-aggregation h)
//   MODE 1: bf16 dense store, +bias, ReLU (encoder / decoder layers)
// One wave computes a 16-row strip across all NT column tiles; A fragment is
// reused across tiles; B tiles are tiny and L2-resident.
// ---------------------------------------------------------------------------
template <int NT, int MODE>
__global__ __launch_bounds__(256) void gemm_wmma_kernel(
    const bf16* __restrict__ A, int lda,
    const bf16* __restrict__ Wt, int ldw,
    const float* __restrict__ bias,
    void* __restrict__ Out, int M, int K)
{
    const int wave = blockIdx.x * (blockDim.x >> 5) + (threadIdx.x >> 5);
    const int m0 = wave * 16;
    if (m0 >= M) return;                       // wave-uniform; EXEC stays all-1s
    const int lane = threadIdx.x & 31;
    const int half = lane >> 4;                // lane group
    const int r16  = lane & 15;

    v8f acc[NT];
#pragma unroll
    for (int t = 0; t < NT; ++t) { v8f z = {}; acc[t] = z; }

    // A fragment base: 16-bit A 16x32 layout -> lane half 0 holds K {0..7,16..23},
    // half 1 holds K {8..15,24..31}, row = lane&15.
    const bf16* arow = A + (size_t)(m0 + r16) * lda + half * 8;

    for (int k0 = 0; k0 < K; k0 += 32) {
        v8bf alo = *(const v8bf*)(arow + k0);
        v8bf ahi = *(const v8bf*)(arow + k0 + 16);
        v16bf a = __builtin_shufflevector(alo, ahi,
                    0,1,2,3,4,5,6,7,8,9,10,11,12,13,14,15);
#pragma unroll
        for (int t = 0; t < NT; ++t) {
            // B fragment: same K order as A -> lanes 0-15 hold K {0..7,16..23},
            // lanes 16-31 hold K {8..15,24..31}, column = lane&15. With Wt stored
            // [N][K] row-major this is two 8-element reads.
            const bf16* brow = Wt + (size_t)(t * 16 + r16) * ldw + k0 + half * 8;
            v8bf blo = *(const v8bf*)(brow);
            v8bf bhi = *(const v8bf*)(brow + 16);
            v16bf b = __builtin_shufflevector(blo, bhi,
                        0,1,2,3,4,5,6,7,8,9,10,11,12,13,14,15);
            acc[t] = __builtin_amdgcn_wmma_f32_16x16x32_bf16(
                        false, a, false, b, (short)0, acc[t], false, false);
        }
    }

    // C/D layout: VGPR r, lanes 0-15 -> M = r; lanes 16-31 -> M = 8 + r; N = lane&15
    const int Nout = NT * 16;
    const int row_base = m0 + half * 8;
    if (MODE == 0) {
        float* out = (float*)Out;
#pragma unroll
        for (int t = 0; t < NT; ++t) {
            const int col = t * 16 + r16;
#pragma unroll
            for (int r = 0; r < 8; ++r)
                out[(size_t)(row_base + r) * Nout + col] = acc[t][r];
        }
    } else {
        bf16* out = (bf16*)Out;
#pragma unroll
        for (int t = 0; t < NT; ++t) {
            const int col = t * 16 + r16;
            const float bb = bias[col];
#pragma unroll
            for (int r = 0; r < 8; ++r) {
                float v = acc[t][r] + bb;
                v = v > 0.f ? v : 0.f;
                out[(size_t)(row_base + r) * Nout + col] = (bf16)v;
            }
        }
    }
}

// ---------------------------------------------------------------------------
// Weight convert + transpose: W[fi][fo] f32 -> Wt[fo][Kpad] bf16, zero padded
// ---------------------------------------------------------------------------
__global__ __launch_bounds__(256) void wconv_kernel(
    const float* __restrict__ W, bf16* __restrict__ Wt, int fi, int fo, int Kpad)
{
    int idx = blockIdx.x * blockDim.x + threadIdx.x;
    if (idx >= fo * Kpad) return;
    int n = idx / Kpad, k = idx - n * Kpad;
    float v = (k < fi) ? W[(size_t)k * fo + n] : 0.f;
    Wt[idx] = (bf16)v;
}

// Feature assembly: [local_x[voxel_type[i]], voxel_x[i], label_hard[i], 0-pad]
__global__ __launch_bounds__(256) void assemble_kernel(
    const float* __restrict__ local_x, const float* __restrict__ voxel_x,
    const float* __restrict__ label_hard, const int* __restrict__ vtype,
    bf16* __restrict__ out)
{
    long long idx = (long long)blockIdx.x * blockDim.x + threadIdx.x;
    if (idx >= (long long)NN * INPAD) return;
    int i = (int)(idx / INPAD), c = (int)(idx - (long long)i * INPAD);
    float v;
    if (c < LDIM)               v = local_x[(size_t)vtype[i] * LDIM + c];
    else if (c < LDIM + VDIM)   v = voxel_x[(size_t)i * VDIM + (c - LDIM)];
    else if (c < INDIM)         v = label_hard[(size_t)i * NCLS + (c - LDIM - VDIM)];
    else                        v = 0.f;
    out[idx] = (bf16)v;
}

__global__ __launch_bounds__(256) void fill_f32_kernel(float* __restrict__ p, float v, int n)
{
    int i = blockIdx.x * blockDim.x + threadIdx.x;
    if (i < n) p[i] = v;
}

__global__ __launch_bounds__(256) void deg_kernel(const int* __restrict__ ei, float* __restrict__ deg)
{
    int e = blockIdx.x * blockDim.x + threadIdx.x;
    if (e < NE) atomicAdd(&deg[ei[NE + e]], 1.0f);
}

__global__ __launch_bounds__(256) void dis_kernel(const float* __restrict__ deg, float* __restrict__ dis)
{
    int i = blockIdx.x * blockDim.x + threadIdx.x;
    if (i < NN) dis[i] = rsqrtf(deg[i]);     // deg >= 1 due to self loops
}

// agg[i][f] = bias[f] + h[i][f]*dis[i]^2  (self loop folded into init)
__global__ __launch_bounds__(256) void agg_init_kernel(
    const float* __restrict__ h, const float* __restrict__ bias,
    const float* __restrict__ dis, float* __restrict__ agg, int fo)
{
    long long idx = (long long)blockIdx.x * blockDim.x + threadIdx.x;
    if (idx >= (long long)NN * fo) return;
    int i = (int)(idx / fo), f = (int)(idx - (long long)i * fo);
    float d = dis[i];
    agg[idx] = bias[f] + h[idx] * d * d;
}

// scatter-add h[src]*norm into agg[dst]; one thread per (edge, 4 features)
__global__ __launch_bounds__(256) void scatter_kernel(
    const float* __restrict__ h, const int* __restrict__ ei,
    const float* __restrict__ dis, float* __restrict__ agg, int fo)
{
    const int per = fo >> 2;
    long long idx = (long long)blockIdx.x * blockDim.x + threadIdx.x;
    if (idx >= (long long)NE * per) return;
    int e = (int)(idx / per), g = (int)(idx - (long long)e * per);
    int s = ei[e], d = ei[NE + e];
    float nrm = dis[s] * dis[d];
    const float4 hv = *(const float4*)(h + (size_t)s * fo + g * 4);
    float* ag = agg + (size_t)d * fo + g * 4;
    atomicAdd(ag + 0, hv.x * nrm);
    atomicAdd(ag + 1, hv.y * nrm);
    atomicAdd(ag + 2, hv.z * nrm);
    atomicAdd(ag + 3, hv.w * nrm);
}

__global__ __launch_bounds__(256) void ln_stats_kernel(
    const float* __restrict__ x, long long total, float* __restrict__ stats)
{
    __shared__ float s1[256], s2[256];
    float a = 0.f, b = 0.f;
    for (long long i = (long long)blockIdx.x * blockDim.x + threadIdx.x;
         i < total; i += (long long)gridDim.x * blockDim.x) {
        float v = x[i]; a += v; b += v * v;
    }
    s1[threadIdx.x] = a; s2[threadIdx.x] = b;
    __syncthreads();
    for (int off = 128; off > 0; off >>= 1) {
        if ((int)threadIdx.x < off) {
            s1[threadIdx.x] += s1[threadIdx.x + off];
            s2[threadIdx.x] += s2[threadIdx.x + off];
        }
        __syncthreads();
    }
    if (threadIdx.x == 0) { atomicAdd(&stats[0], s1[0]); atomicAdd(&stats[1], s2[0]); }
}

// graph-wide LN + affine + ReLU -> bf16 activations (dense stride fo)
__global__ __launch_bounds__(256) void ln_apply_kernel(
    const float* __restrict__ agg, const float* __restrict__ stats,
    const float* __restrict__ gam, const float* __restrict__ bet,
    bf16* __restrict__ out, int fo, float inv_count)
{
    long long idx = (long long)blockIdx.x * blockDim.x + threadIdx.x;
    if (idx >= (long long)NN * fo) return;
    int f = (int)(idx % fo);
    float mu  = stats[0] * inv_count;
    float var = stats[1] * inv_count - mu * mu;
    float sd  = sqrtf(var > 0.f ? var : 0.f);
    float v = (agg[idx] - mu) / (sd + EPS_LN) * gam[f] + bet[f];
    v = v > 0.f ? v : 0.f;
    out[idx] = (bf16)v;
}

// final head: sigmoid(x[16] . W + b)
__global__ __launch_bounds__(256) void dec_final_kernel(
    const bf16* __restrict__ x, const float* __restrict__ W,
    const float* __restrict__ b, float* __restrict__ out)
{
    int i = blockIdx.x * blockDim.x + threadIdx.x;
    if (i >= NN) return;
    float s = b[0];
#pragma unroll
    for (int f = 0; f < 16; ++f) s += (float)x[(size_t)i * 16 + f] * W[f];
    out[i] = 1.f / (1.f + __expf(-s));
}

__global__ void zero8_kernel(float* __restrict__ p)
{
    if (threadIdx.x < 8) p[threadIdx.x] = 0.f;
}

// ---------------------------------------------------------------------------
static void launch_gemm(int NT, int MODE, const bf16* A, int lda, const bf16* Wt,
                        int ldw, const float* bias, void* out, int M, int K,
                        hipStream_t s)
{
    int blocks = ((M + 15) / 16 + 7) / 8;
    dim3 g(blocks), b(256);
#define CASE(nt, md) if (NT == nt && MODE == md) { \
    gemm_wmma_kernel<nt, md><<<g, b, 0, s>>>(A, lda, Wt, ldw, bias, out, M, K); return; }
    CASE(1,0) CASE(1,1) CASE(2,0) CASE(2,1)
    CASE(4,0) CASE(4,1) CASE(8,0) CASE(8,1)
#undef CASE
}

extern "C" void kernel_launch(void* const* d_in, const int* in_sizes, int n_in,
                              void* d_out, int out_size, void* d_ws, size_t ws_size,
                              hipStream_t stream)
{
    (void)in_sizes; (void)n_in; (void)out_size; (void)ws_size;
    const float* local_x    = (const float*)d_in[0];
    const float* voxel_x    = (const float*)d_in[1];
    const int*   voxel_type = (const int*)  d_in[2];
    const int*   edge_index = (const int*)  d_in[3];
    const float* label_hard = (const float*)d_in[4];
    // params flattened in pytree order: convs(W,b,g,beta)x4, dec(W,b)x4, enc(W,b)x2
    const float* cW[4]; const float* cB[4]; const float* cG[4]; const float* cBe[4];
    for (int i = 0; i < 4; ++i) {
        cW[i]  = (const float*)d_in[5 + 4 * i + 0];
        cB[i]  = (const float*)d_in[5 + 4 * i + 1];
        cG[i]  = (const float*)d_in[5 + 4 * i + 2];
        cBe[i] = (const float*)d_in[5 + 4 * i + 3];
    }
    const float* dW[4]; const float* dB[4];
    for (int i = 0; i < 4; ++i) {
        dW[i] = (const float*)d_in[21 + 2 * i + 0];
        dB[i] = (const float*)d_in[21 + 2 * i + 1];
    }
    const float* eW[2]; const float* eB[2];
    for (int i = 0; i < 2; ++i) {
        eW[i] = (const float*)d_in[29 + 2 * i + 0];
        eB[i] = (const float*)d_in[29 + 2 * i + 1];
    }

    // ---- workspace layout (bytes); region A shared between act0 and hbuf ----
    char* ws = (char*)d_ws;
    const size_t N = NN;
    bf16*  act0 = (bf16*)(ws + 0);                       // N x 224 bf16
    float* hbuf = (float*)(ws + 0);                      // N x 128 f32 (after enc)
    float* agg  = (float*)(ws + N * 512);                // N x 128 f32
    bf16*  act1 = (bf16*)(ws + N * 512 * 2);             // N x 128 bf16
    bf16*  act2 = (bf16*)(ws + N * 512 * 2 + N * 256);   // N x 128 bf16
    float* deg  = (float*)(ws + N * 512 * 2 + N * 512);
    float* dis  = deg + N;
    bf16*  wbase = (bf16*)((char*)(dis + N) + 256);
    // per-layer bf16 transposed weights
    bf16* w_enc0 = wbase;                 // 128 x 224
    bf16* w_enc1 = w_enc0 + 128 * 224;    // 128 x 128
    bf16* w_cv[4];
    w_cv[0] = w_enc1 + 128 * 128;         // 64 x 128
    w_cv[1] = w_cv[0] + 64 * 128;         // 32 x 64
    w_cv[2] = w_cv[1] + 32 * 64;          // 64 x 32
    w_cv[3] = w_cv[2] + 64 * 32;          // 128 x 64
    bf16* w_dc[3];
    w_dc[0] = w_cv[3] + 128 * 64;         // 64 x 128
    w_dc[1] = w_dc[0] + 64 * 128;         // 32 x 64
    w_dc[2] = w_dc[1] + 32 * 64;          // 16 x 32
    float* stats = (float*)((char*)(w_dc[2] + 16 * 32) + 256); // 8 f32

    const int cfi[4] = {128, 64, 32, 64};
    const int cfo[4] = {64, 32, 64, 128};

    // ---- weight conversion ----
    auto wc = [&](const float* W, bf16* Wt, int fi, int fo, int Kpad) {
        int n = fo * Kpad;
        wconv_kernel<<<(n + 255) / 256, 256, 0, stream>>>(W, Wt, fi, fo, Kpad);
    };
    wc(eW[0], w_enc0, INDIM, 128, INPAD);
    wc(eW[1], w_enc1, 128, 128, 128);
    for (int i = 0; i < 4; ++i) wc(cW[i], w_cv[i], cfi[i], cfo[i], cfi[i]);
    wc(dW[0], w_dc[0], 128, 64, 128);
    wc(dW[1], w_dc[1], 64, 32, 64);
    wc(dW[2], w_dc[2], 32, 16, 32);
    zero8_kernel<<<1, 32, 0, stream>>>(stats);

    // ---- degrees / normalization ----
    fill_f32_kernel<<<(NN + 255) / 256, 256, 0, stream>>>(deg, 1.0f, NN);
    deg_kernel<<<(NE + 255) / 256, 256, 0, stream>>>(edge_index, deg);
    dis_kernel<<<(NN + 255) / 256, 256, 0, stream>>>(deg, dis);

    // ---- feature assembly + encoder ----
    {
        long long tot = (long long)NN * INPAD;
        assemble_kernel<<<(int)((tot + 255) / 256), 256, 0, stream>>>(
            local_x, voxel_x, label_hard, voxel_type, act0);
    }
    launch_gemm(8, 1, act0, INPAD, w_enc0, INPAD, eB[0], act1, NN, INPAD, stream);
    launch_gemm(8, 1, act1, 128,   w_enc1, 128,   eB[1], act2, NN, 128,   stream);

    // ---- GCN conv layers ----
    bf16* aIn = act2; bf16* aOut = act1;
    for (int i = 0; i < 4; ++i) {
        const int fi = cfi[i], fo = cfo[i];
        launch_gemm(fo / 16, 0, aIn, fi, w_cv[i], fi, nullptr, hbuf, NN, fi, stream);
        long long nf = (long long)NN * fo;
        agg_init_kernel<<<(int)((nf + 255) / 256), 256, 0, stream>>>(hbuf, cB[i], dis, agg, fo);
        long long ef = (long long)NE * (fo / 4);
        scatter_kernel<<<(int)((ef + 255) / 256), 256, 0, stream>>>(hbuf, edge_index, dis, agg, fo);
        ln_stats_kernel<<<2048, 256, 0, stream>>>(agg, nf, stats + 2 * i);
        ln_apply_kernel<<<(int)((nf + 255) / 256), 256, 0, stream>>>(
            agg, stats + 2 * i, cG[i], cBe[i], aOut, fo, 1.0f / (float)nf);
        bf16* t = aIn; aIn = aOut; aOut = t;
    }

    // ---- decoder ---- (aIn now holds conv output, 128 features)
    launch_gemm(4, 1, aIn, 128, w_dc[0], 128, dB[0], aOut, NN, 128, stream);
    launch_gemm(2, 1, aOut, 64, w_dc[1], 64,  dB[1], aIn,  NN, 64,  stream);
    launch_gemm(1, 1, aIn, 32,  w_dc[2], 32,  dB[2], aOut, NN, 32,  stream);
    dec_final_kernel<<<(NN + 255) / 256, 256, 0, stream>>>(aOut, dW[3], dB[3], (float*)d_out);
}
